// relational_LSTM_81896436400337
// MI455X (gfx1250) — matrix operand order repaired and checked by reading, not compiled, not run on hardware
//
#include <hip/hip_runtime.h>
#include <hip/hip_bf16.h>

// ---------------------------------------------------------------------------
// MI455X (gfx1250) implementation of the relational LSTM reference.
// Strategy: bf16 V_WMMA_F32_16X16X32_BF16 for every GEMM; hard top-2 gating
// folded into A-fragment bitmasks + pre-scaled (x0.5) candidate weights;
// recurrence handled by 4 persistent workgroups (one per 16-batch slice) with
// LDS-resident h-state (wave32 / 320KB-LDS WGP model).
// ---------------------------------------------------------------------------

typedef __bf16 bf16;
typedef __attribute__((ext_vector_type(16))) __bf16 v16bf;
typedef __attribute__((ext_vector_type(8)))  float  v8f;

constexpr int nB = 64, nT = 128, nM = 4, nD = 512, nH = 512, nK = 6;
constexpr int nFH = 2560;  // 5*H

union FragU { v16bf v; uint4 u[2]; unsigned int w[8]; };

// A-fragment (16x32 bf16, row-major source, lane holds row lane&15):
//   lanes 0-15: K 0..7 and 16..23 ; lanes 16-31: K 8..15 and 24..31
// caller pre-offsets pointer by khA = (lane<16?0:8) elements.
__device__ __forceinline__ v16bf ld_a16(const bf16* p) {
  FragU f;
  f.u[0] = *(const uint4*)(p);
  f.u[1] = *(const uint4*)(p + 16);
  return f.v;
}
// B-fragment (32x16 bf16 from W^T [N][K]): lane holds column lane&15,
// K khB..khB+7 and 16+khB..16+khB+7, khB = (lane<16?0:8). Caller pre-offsets.
__device__ __forceinline__ v16bf ld_b16(const bf16* p) {
  FragU f;
  f.u[0] = *(const uint4*)(p);
  f.u[1] = *(const uint4*)(p + 16);
  return f.v;
}
__device__ __forceinline__ v16bf zero_frag() {
  FragU f;
#pragma unroll
  for (int i = 0; i < 8; ++i) f.w[i] = 0u;
  return f.v;
}
__device__ __forceinline__ v16bf and_mask(v16bf a, unsigned m) {
  FragU f; f.v = a;
#pragma unroll
  for (int i = 0; i < 8; ++i) f.w[i] &= m;
  return f.v;
}
__device__ __forceinline__ v8f zero_acc() {
  v8f a = {0.f, 0.f, 0.f, 0.f, 0.f, 0.f, 0.f, 0.f};
  return a;
}
__device__ __forceinline__ v8f wmma_bf16(v16bf a, v16bf b, v8f c) {
  return __builtin_amdgcn_wmma_f32_16x16x32_bf16(false, a, false, b,
                                                 (short)0, c, false, false);
}
__device__ __forceinline__ float sigm(float x) { return 1.f / (1.f + __expf(-x)); }

// ---------------------------------------------------------------------------
// prep kernels
// ---------------------------------------------------------------------------
__global__ void cvt_bf16_kernel(const float* __restrict__ src,
                                bf16* __restrict__ dst, long n) {
  long i = (long)blockIdx.x * blockDim.x + threadIdx.x;
  if (i < n) dst[i] = (bf16)src[i];
}

// src [batch][K][N] f32 -> dst [batch][N][K] bf16 (scaled)
__global__ void transpose_bf16_kernel(const float* __restrict__ src,
                                      bf16* __restrict__ dst,
                                      int K, int N, float scale) {
  long mat = (long)K * N;
  const float* s = src + (long)blockIdx.z * mat;
  bf16* d = dst + (long)blockIdx.z * mat;
  long i = (long)blockIdx.x * blockDim.x + threadIdx.x;
  if (i < mat) {
    int k = (int)(i / N), n = (int)(i % N);
    d[(long)n * K + k] = (bf16)(s[i] * scale);
  }
}

// context [B][T][H] f32 -> scene [T][B][H] bf16
__global__ void scene_cvt_kernel(const float* __restrict__ ctx,
                                 bf16* __restrict__ scn) {
  long i = (long)blockIdx.x * blockDim.x + threadIdx.x;
  long n = (long)nB * nT * nH;
  if (i >= n) return;
  int h = (int)(i % nH);
  long bt = i / nH;
  int t = (int)(bt % nT);
  int b = (int)(bt / nT);
  scn[((long)t * nB + b) * nH + h] = (bf16)ctx[i];
}

__global__ void bias_add_kernel(const float* __restrict__ a,
                                const float* __restrict__ b,
                                float* __restrict__ o, int n) {
  int i = blockIdx.x * blockDim.x + threadIdx.x;
  if (i < n) o[i] = a[i] + b[i];
}

// ---------------------------------------------------------------------------
// share_feat = relu(A[M,K] @ W[K,N] + bias)  (A bf16 row-major, BT = [N][K])
// block = 128 threads (4 waves); each wave -> 16 rows x 32 cols
// ---------------------------------------------------------------------------
__global__ __launch_bounds__(128) void gemm_bias_relu_kernel(
    const bf16* __restrict__ A, const bf16* __restrict__ BT,
    const float* __restrict__ bias, float* __restrict__ C,
    int M, int N, int Kdim) {
  const int wave = threadIdx.x >> 5;
  const int lane = threadIdx.x & 31;
  const int m0 = blockIdx.x * 16;
  const int n0 = blockIdx.y * 128 + wave * 32;
  const int r = lane & 15;
  const int khA = (lane < 16) ? 0 : 8;
  const int khB = (lane < 16) ? 0 : 8;
  const bf16* Ap = A + (long)(m0 + r) * Kdim + khA;
  const bf16* Bp0 = BT + (long)(n0 + (lane & 15)) * Kdim + khB;
  const bf16* Bp1 = Bp0 + (long)16 * Kdim;
  v8f acc0 = zero_acc(), acc1 = zero_acc();
  for (int k0 = 0; k0 < Kdim; k0 += 32) {
    v16bf a = ld_a16(Ap + k0);
    acc0 = wmma_bf16(a, ld_b16(Bp0 + k0), acc0);
    acc1 = wmma_bf16(a, ld_b16(Bp1 + k0), acc1);
  }
  const int rb = (lane < 16) ? 0 : 8;
  const int col = lane & 15;
#pragma unroll
  for (int rr = 0; rr < 8; ++rr) {
    const int row = m0 + rb + rr;
    float v0 = acc0[rr] + bias[n0 + col];
    float v1 = acc1[rr] + bias[n0 + 16 + col];
    C[(long)row * N + n0 + col] = fmaxf(v0, 0.f);
    C[(long)row * N + n0 + 16 + col] = fmaxf(v1, 0.f);
  }
}

// ---------------------------------------------------------------------------
// hard group separation -> hid0 [T][2][B][D] bf16
// ---------------------------------------------------------------------------
__global__ void group_sep_kernel(const float* __restrict__ feat,
                                 const float* __restrict__ sepW,
                                 const float* __restrict__ sepb,
                                 bf16* __restrict__ hid0) {
  const int bt = blockIdx.x;
  const int b = bt >> 7, t = bt & 127;
  const float* F = feat + (long)bt * (nM * nD);
  __shared__ float red[nM * 256];
  const int tid = threadIdx.x;
  float p[nM];
#pragma unroll
  for (int m = 0; m < nM; ++m) p[m] = 0.f;
  for (int d = tid; d < nD; d += 256) {
    float w = sepW[d];
#pragma unroll
    for (int m = 0; m < nM; ++m) p[m] += F[m * nD + d] * w;
  }
#pragma unroll
  for (int m = 0; m < nM; ++m) red[m * 256 + tid] = p[m];
  __syncthreads();
  for (int s = 128; s > 0; s >>= 1) {
    if (tid < s) {
#pragma unroll
      for (int m = 0; m < nM; ++m) red[m * 256 + tid] += red[m * 256 + tid + s];
    }
    __syncthreads();
  }
  float sc[nM], mx = -3.4e38f;
#pragma unroll
  for (int m = 0; m < nM; ++m) { sc[m] = red[m * 256] + sepb[0]; mx = fmaxf(mx, sc[m]); }
  float es = 0.f;
#pragma unroll
  for (int m = 0; m < nM; ++m) { sc[m] = __expf(sc[m] - mx); es += sc[m]; }
  float sw[nM], c0 = 0.f;
#pragma unroll
  for (int m = 0; m < nM; ++m) { sw[m] = ((sc[m] / es - 0.25f) > 0.f) ? 1.f : 0.f; c0 += sw[m]; }
  const float inv0 = 1.f / (c0 + 1e-8f);
  const float inv1 = 1.f / ((4.f - c0) + 1e-8f);
  for (int d = tid; d < nD; d += 256) {
    float g0 = 0.f, g1 = 0.f;
#pragma unroll
    for (int m = 0; m < nM; ++m) {
      float v = F[m * nD + d];
      g0 += sw[m] * v;
      g1 += (1.f - sw[m]) * v;
    }
    hid0[(((long)t * 2 + 0) * nB + b) * nD + d] = (bf16)(g0 * inv0);
    hid0[(((long)t * 2 + 1) * nB + b) * nD + d] = (bf16)(g1 * inv1);
  }
}

// ---------------------------------------------------------------------------
// Recurrent scan: 4 persistent workgroups x 512 threads (16 waves).
// WG owns 16 batches x 2 groups = two 16-row WMMA tiles. h-state in LDS bf16
// double buffer; gates staged in LDS bf16; cell state in private global slab.
// Wave w: j = w/8, column block nb = w%8 (320 cols = 20 N-tiles, 10x2 groups).
// ---------------------------------------------------------------------------
__global__ __launch_bounds__(512) void scan_kernel(
    const bf16* __restrict__ hid_in,    // [T][2][B][D]
    const bf16* __restrict__ scn,       // [T][B][H]
    const float* __restrict__ share_in, // [B][T][H]
    const bf16* __restrict__ WxT,       // [2][FH][H]
    const bf16* __restrict__ WhT,       // [2][FH][H]
    const bf16* __restrict__ WcT,       // [2][K][FH][H]  (pre-scaled by 0.5)
    const float* __restrict__ aW,       // [H][K]
    const float* __restrict__ ab,       // [K]
    const float* __restrict__ bias0,    // [2][FH]  (= bx + bh)
    const float* __restrict__ bc,       // [2][K][FH]
    float* __restrict__ cell_ws,        // [4][2][16][H]
    bf16* __restrict__ hid_out_bf,      // [T][2][B][D]
    float* __restrict__ hid_out_f,      // [B][T][2][D]
    float* __restrict__ share_out) {    // [B][T][H]
  extern __shared__ char smem[];
  bf16* hbuf  = (bf16*)smem;                        // [2][2][16][512] 64KB
  bf16* gates = (bf16*)(smem + 65536);              // [2][16][2560] 160KB
  float* aw_l = (float*)(smem + 65536 + 163840);    // [512*6]
  float* ab_l = aw_l + 512 * 6;                     // [8]
  float* lgts = ab_l + 8;                           // [2*16*6]
  float* wgt  = lgts + 192;                         // [2*16*6]
  int* tk     = (int*)(wgt + 192);                  // [2*16*2]
  int* kuse   = tk + 64;                            // [2]

  const int tid = threadIdx.x;
  const int wgid = blockIdx.x;  // 0..3
  const int b0 = wgid * 16;

  for (int i = tid; i < 512 * 6; i += 512) aw_l[i] = aW[i];
  if (tid < 6) ab_l[tid] = ab[tid];
  for (int i = tid; i < 2 * 2 * 16 * 512; i += 512) hbuf[i] = (bf16)0.f;
  float* cellp = cell_ws + (long)wgid * 2 * 16 * 512;
  for (int i = tid; i < 2 * 16 * 512; i += 512) cellp[i] = 0.f;
  float sharer[16], cumr[16];
#pragma unroll
  for (int b = 0; b < 16; ++b) { sharer[b] = 0.f; cumr[b] = 0.f; }
  __syncthreads();

  const int wave = tid >> 5;
  const int lane = tid & 31;
  const int jw = wave >> 3;
  const int nb = wave & 7;
  const int r = lane & 15;
  const int khA = (lane < 16) ? 0 : 8;
  const int khB = (lane < 16) ? 0 : 8;
  const int nlan = lane & 15;

  for (int t = 0; t < nT; ++t) {
    const int s1 = t & 1;   // h_{t-1} slot
    const int s2 = s1 ^ 1;  // h_{t-2} slot (also h_t destination)

    // ---- Phase A: router logits + hard top-2 ----
    if (tid < 2) kuse[tid] = 0;
    if (tid < 192) {
      int j = tid / 96, b = (tid / 6) % 16, k = tid % 6;
      const bf16* hp = hbuf + ((s1 * 2 + j) * 16 + b) * 512;
      float acc = ab_l[k];
      for (int h = 0; h < 512; ++h) acc += (float)hp[h] * aw_l[h * 6 + k];
      lgts[(j * 16 + b) * 6 + k] = acc;
    }
    __syncthreads();
    if (tid < 32) {
      int j = tid >> 4, b = tid & 15;
      float* L = lgts + (j * 16 + b) * 6;
      int i0 = 0; float v0 = L[0];
#pragma unroll
      for (int k = 1; k < 6; ++k) if (L[k] > v0) { v0 = L[k]; i0 = k; }
      int i1 = -1; float v1 = 0.f;
#pragma unroll
      for (int k = 0; k < 6; ++k) {
        if (k == i0) continue;
        if (i1 < 0 || L[k] > v1) { v1 = L[k]; i1 = k; }
      }
      float* W = wgt + (j * 16 + b) * 6;
#pragma unroll
      for (int k = 0; k < 6; ++k) W[k] = 0.f;
      W[i0] = 0.5f; W[i1] = 0.5f;
      tk[(j * 16 + b) * 2 + 0] = i0;
      tk[(j * 16 + b) * 2 + 1] = i1;
      atomicOr(&kuse[j], (1 << i0) | (1 << i1));
    }
    __syncthreads();

    // ---- Phase B: all gate GEMMs via WMMA ----
    {
      unsigned mk[6];
      const float* Wr = wgt + (jw * 16 + r) * 6;
#pragma unroll
      for (int k = 0; k < 6; ++k) mk[k] = (Wr[k] != 0.f) ? 0xFFFFFFFFu : 0u;
      const int ku = kuse[jw];  // wave-uniform: skip unused candidate GEMMs
      const int jo = 1 - jw;
      const bool hasN = (t + 1 < nT);
      const bf16* seqP  = hid_in + (((long)t * 2 + jw) * nB + b0 + r) * 512 + khA;
      const bf16* seqPo = hid_in + (((long)t * 2 + jo) * nB + b0 + r) * 512 + khA;
      const bf16* sqnP  = hid_in + (((long)(t + 1) * 2 + jw) * nB + b0 + r) * 512 + khA;
      const bf16* sqnPo = hid_in + (((long)(t + 1) * 2 + jo) * nB + b0 + r) * 512 + khA;
      const bf16* scnP  = scn + ((long)t * nB + b0 + r) * 512 + khA;
      const bf16* h1P  = hbuf + ((s1 * 2 + jw) * 16 + r) * 512 + khA;
      const bf16* h1Po = hbuf + ((s1 * 2 + jo) * 16 + r) * 512 + khA;
      const bf16* h2P  = hbuf + ((s2 * 2 + jw) * 16 + r) * 512 + khA;
      const bf16* BxB = WxT + (long)jw * nFH * 512;
      const bf16* BhB = WhT + (long)jw * nFH * 512;
      const bf16* BcB = WcT + (long)jw * nK * nFH * 512;
      for (int ng = 0; ng < 10; ++ng) {
        v8f acc[2];
        acc[0] = zero_acc(); acc[1] = zero_acc();
        for (int kk = 0; kk < 16; ++kk) {
          const int k0 = kk * 32;
          v16bf aSeq = ld_a16(seqP + k0);
          v16bf aH1  = ld_a16(h1P + k0);
          v16bf aH2  = and_mask(ld_a16(h2P + k0), mk[0]);
          v16bf aH1o = and_mask(ld_a16(h1Po + k0), mk[2]);
          v16bf aSqo = and_mask(ld_a16(seqPo + k0), mk[3]);
          v16bf aScn = and_mask(ld_a16(scnP + k0), mk[5]);
          v16bf aSqn, aSqno;
          if (hasN) {
            aSqn  = and_mask(ld_a16(sqnP + k0), mk[1]);
            aSqno = and_mask(ld_a16(sqnPo + k0), mk[4]);
          } else {
            aSqn = zero_frag(); aSqno = zero_frag();
          }
#pragma unroll
          for (int q = 0; q < 2; ++q) {
            const int n0 = (nb * 20 + ng * 2 + q) * 16;
            const long boff = ((long)(n0 + nlan)) * 512 + k0 + khB;
            acc[q] = wmma_bf16(aSeq, ld_b16(BxB + boff), acc[q]);
            acc[q] = wmma_bf16(aH1, ld_b16(BhB + boff), acc[q]);
            if (ku & 1)  acc[q] = wmma_bf16(aH2,  ld_b16(BcB + (long)0 * nFH * 512 + boff), acc[q]);
            if (ku & 2)  acc[q] = wmma_bf16(aSqn, ld_b16(BcB + (long)1 * nFH * 512 + boff), acc[q]);
            if (ku & 4)  acc[q] = wmma_bf16(aH1o, ld_b16(BcB + (long)2 * nFH * 512 + boff), acc[q]);
            if (ku & 8)  acc[q] = wmma_bf16(aSqo, ld_b16(BcB + (long)3 * nFH * 512 + boff), acc[q]);
            if (ku & 16) acc[q] = wmma_bf16(aSqno, ld_b16(BcB + (long)4 * nFH * 512 + boff), acc[q]);
            if (ku & 32) acc[q] = wmma_bf16(aScn, ld_b16(BcB + (long)5 * nFH * 512 + boff), acc[q]);
          }
        }
        const int rb = (lane < 16) ? 0 : 8;
#pragma unroll
        for (int q = 0; q < 2; ++q) {
          const int n0 = (nb * 20 + ng * 2 + q) * 16 + nlan;
#pragma unroll
          for (int rr = 0; rr < 8; ++rr) {
            gates[(jw * 16 + rb + rr) * nFH + n0] = (bf16)acc[q][rr];
          }
        }
      }
    }
    __syncthreads();

    // ---- Phase C: LSTM pointwise update; thread tid owns column c = tid ----
    {
      const int c = tid;
#pragma unroll
      for (int b = 0; b < 16; ++b) {
        float sgsum = 0.f;
#pragma unroll
        for (int j = 0; j < 2; ++j) {
          const bf16* G = gates + (j * 16 + b) * nFH;
          const float* B0 = bias0 + (long)j * nFH;
          const int ka = tk[(j * 16 + b) * 2 + 0];
          const int kb = tk[(j * 16 + b) * 2 + 1];
          const float* Bca = bc + ((long)j * nK + ka) * nFH;
          const float* Bcb = bc + ((long)j * nK + kb) * nFH;
          float ig = (float)G[c]        + B0[c]        + 0.5f * (Bca[c]        + Bcb[c]);
          float fg = (float)G[512 + c]  + B0[512 + c]  + 0.5f * (Bca[512 + c]  + Bcb[512 + c]);
          float og = (float)G[1024 + c] + B0[1024 + c] + 0.5f * (Bca[1024 + c] + Bcb[1024 + c]);
          float sg = (float)G[1536 + c] + B0[1536 + c] + 0.5f * (Bca[1536 + c] + Bcb[1536 + c]);
          float cg = (float)G[2048 + c] + B0[2048 + c] + 0.5f * (Bca[2048 + c] + Bcb[2048 + c]);
          float* cp = cellp + (j * 16 + b) * 512 + c;
          float cold = *cp;
          float cnew = (1.f - sigm(fg)) * cold + sigm(ig) * tanhf(cg);
          *cp = cnew;
          float h = sigm(og) * tanhf(cnew);
          sgsum += sg;
          hbuf[((s2 * 2 + j) * 16 + b) * 512 + c] = (bf16)h;
          hid_out_bf[(((long)t * 2 + j) * nB + b0 + b) * 512 + c] = (bf16)h;
          hid_out_f[(((long)(b0 + b) * nT + t) * 2 + j) * 512 + c] = h;
        }
        float gate = sigm(sgsum);
        const long sidx = ((long)(b0 + b) * nT + t) * 512 + c;
        sharer[b] += gate * share_in[sidx];
        cumr[b] += gate;
        share_out[sidx] = sharer[b] / cumr[b];
      }
    }
    __syncthreads();
  }
}

// ---------------------------------------------------------------------------
// out[b,t] = concat( (hid0+hid1)/2 [1024], (share0+share1)/2 [512] )
// ---------------------------------------------------------------------------
__global__ void final_kernel(const float* __restrict__ h0,
                             const float* __restrict__ h1,
                             const float* __restrict__ s0,
                             const float* __restrict__ s1,
                             float* __restrict__ out) {
  long i = (long)blockIdx.x * blockDim.x + threadIdx.x;
  long n = (long)nB * nT * 1536;
  if (i >= n) return;
  int f = (int)(i % 1536);
  long bt = i / 1536;
  if (f < 1024)
    out[i] = 0.5f * (h0[bt * 1024 + f] + h1[bt * 1024 + f]);
  else
    out[i] = 0.5f * (s0[bt * 512 + (f - 1024)] + s1[bt * 512 + (f - 1024)]);
}

// ---------------------------------------------------------------------------
extern "C" void kernel_launch(void* const* d_in, const int* in_sizes, int n_in,
                              void* d_out, int out_size, void* d_ws, size_t ws_size,
                              hipStream_t stream) {
  (void)in_sizes; (void)n_in; (void)out_size; (void)ws_size;
  const float* feat   = (const float*)d_in[0];
  const float* ctx    = (const float*)d_in[1];
  const float* l2s_W  = (const float*)d_in[2];
  const float* l2s_b  = (const float*)d_in[3];
  const float* sep_W  = (const float*)d_in[4];
  const float* sep_b  = (const float*)d_in[5];
  const float* arch_W = (const float*)d_in[6];
  const float* arch_b = (const float*)d_in[7];
  const float* Wx     = (const float*)d_in[8];
  const float* bx     = (const float*)d_in[9];
  const float* Wh     = (const float*)d_in[10];
  const float* bh     = (const float*)d_in[11];
  const float* Wc     = (const float*)d_in[12];
  const float* bc     = (const float*)d_in[13];
  float* out = (float*)d_out;

  char* ws = (char*)d_ws;
  size_t off = 0;
  auto alloc = [&](size_t bytes) -> char* {
    char* p = ws + off;
    off = (off + bytes + 255) & ~(size_t)255;
    return p;
  };
  bf16* l2sT    = (bf16*)alloc((size_t)2048 * 512 * 2);
  bf16* WxT     = (bf16*)alloc((size_t)4 * nFH * 512 * 2);
  bf16* WhT     = (bf16*)alloc((size_t)4 * nFH * 512 * 2);
  bf16* WcT     = (bf16*)alloc((size_t)24 * nFH * 512 * 2);
  bf16* featbf  = (bf16*)alloc((size_t)nB * nT * nM * nD * 2);
  bf16* hidbf0  = (bf16*)alloc((size_t)nT * 2 * nB * nD * 2);
  bf16* hidbf1  = (bf16*)alloc((size_t)nT * 2 * nB * nD * 2);
  bf16* scnbf   = (bf16*)alloc((size_t)nT * nB * nH * 2);
  float* share0 = (float*)alloc((size_t)nB * nT * nH * 4);
  float* shareA = (float*)alloc((size_t)nB * nT * nH * 4);
  float* shareB = (float*)alloc((size_t)nB * nT * nH * 4);
  float* hidf0  = (float*)alloc((size_t)nB * nT * 2 * nD * 4);
  float* hidf1  = (float*)alloc((size_t)nB * nT * 2 * nD * 4);
  float* bias0  = (float*)alloc((size_t)2 * 2 * nFH * 4);
  float* cellws = (float*)alloc((size_t)4 * 2 * 16 * nH * 4);

  // --- weight / activation prep ---
  {
    long n = (long)2048 * 512;
    transpose_bf16_kernel<<<dim3((unsigned)((n + 255) / 256), 1, 1), 256, 0, stream>>>(
        l2s_W, l2sT, 2048, 512, 1.f);
  }
  {
    long n = (long)512 * nFH;
    transpose_bf16_kernel<<<dim3((unsigned)((n + 255) / 256), 1, 4), 256, 0, stream>>>(
        Wx, WxT, 512, nFH, 1.f);
    transpose_bf16_kernel<<<dim3((unsigned)((n + 255) / 256), 1, 4), 256, 0, stream>>>(
        Wh, WhT, 512, nFH, 1.f);
    transpose_bf16_kernel<<<dim3((unsigned)((n + 255) / 256), 1, 24), 256, 0, stream>>>(
        Wc, WcT, 512, nFH, 0.5f);  // fold 1/TOP into candidate weights
  }
  {
    long n = (long)nB * nT * nM * nD;
    cvt_bf16_kernel<<<(unsigned)((n + 255) / 256), 256, 0, stream>>>(feat, featbf, n);
  }
  {
    long n = (long)nB * nT * nH;
    scene_cvt_kernel<<<(unsigned)((n + 255) / 256), 256, 0, stream>>>(ctx, scnbf);
  }
  {
    int n = 2 * 2 * nFH;
    bias_add_kernel<<<(n + 255) / 256, 256, 0, stream>>>(bx, bh, bias0, n);
  }
  // share_feat = relu(feat_flat @ l2s_W + b): M=8192, N=512, K=2048
  gemm_bias_relu_kernel<<<dim3(8192 / 16, 512 / 128), 128, 0, stream>>>(
      featbf, l2sT, l2s_b, share0, 8192, 512, 2048);
  // hard group separation -> hid tiles
  group_sep_kernel<<<nB * nT, 256, 0, stream>>>(feat, sep_W, sep_b, hidbf0);

  // --- recurrent layers (dynamic LDS ~244KB; CDNA5 WGP has 320KB) ---
  const size_t smem_bytes =
      65536 + 163840 + 512 * 6 * 4 + 8 * 4 + 192 * 4 + 192 * 4 + 64 * 4 + 2 * 4 + 256;
  scan_kernel<<<4, 512, smem_bytes, stream>>>(
      hidbf0, scnbf, share0, WxT, WhT, WcT, arch_W, arch_b, bias0, bc, cellws,
      hidbf1, hidf0, shareA);
  scan_kernel<<<4, 512, smem_bytes, stream>>>(
      hidbf1, scnbf, shareA,
      WxT + (long)2 * nFH * 512, WhT + (long)2 * nFH * 512, WcT + (long)12 * nFH * 512,
      arch_W + 512 * 6, arch_b + 6, bias0 + 2 * nFH, bc + (long)2 * nK * nFH, cellws,
      hidbf0, hidf1, shareB);

  {
    long n = (long)nB * nT * 1536;
    final_kernel<<<(unsigned)((n + 255) / 256), 256, 0, stream>>>(
        hidf0, hidf1, shareA, shareB, out);
  }
}
